// RegionGraphGNN_687194767399
// MI455X (gfx1250) — compile-verified, matrix operand order checked by reading, not run on hardware
//
#include <hip/hip_runtime.h>
#include <hip/hip_bf16.h>

#define GN 50000
#define GE 400000
#define CIN 15
#define HID 128
#define HEADS 4
#define MPB 8      // M row-tiles per GEMM block
#define EPW 8      // edges per wave in GAT scatter

typedef __attribute__((ext_vector_type(16))) __bf16 v16bf;
typedef __attribute__((ext_vector_type(8)))  float  v8f;

// ---------- helpers: order-preserving float<->uint for atomicMax ----------
__device__ __forceinline__ unsigned fenc(float f) {
    unsigned u = __float_as_uint(f);
    return (u & 0x80000000u) ? ~u : (u | 0x80000000u);
}
__device__ __forceinline__ float fdec(unsigned u) {
    unsigned b = (u & 0x80000000u) ? (u & 0x7FFFFFFFu) : ~u;
    return __uint_as_float(b);
}

// ---------- zero fill ----------
__global__ void k_zero(float* __restrict__ p, int n) {
    int i = blockIdx.x * blockDim.x + threadIdx.x;
    if (i < n) p[i] = 0.0f;
}

// ---------- pre-swizzle weight matrix into WMMA B-fragment layout (bf16) ----
// F layout: [ntile][kstep][lane][16 bf16], one thread per (ntile,kstep,lane).
__global__ void k_bfrag(const float* __restrict__ B, __bf16* __restrict__ F,
                        int K, int Ncols, int ntiles) {
    int ksteps = K >> 5;
    int total = ntiles * ksteps * 32;
    int t = blockIdx.x * blockDim.x + threadIdx.x;
    if (t >= total) return;
    int lane = t & 31;
    int ks = (t >> 5) % ksteps;
    int nt = t / (32 * ksteps);
    bool hi = lane >= 16;
    int ml = lane & 15;
    int col = nt * 16 + ml;
    int kb1 = hi ? 8 : 0, kb2 = hi ? 24 : 16;
    __bf16* dst = F + (size_t)t * 16;
    bool cok = col < Ncols;
#pragma unroll
    for (int e = 0; e < 8; ++e) {
        int k1 = ks * 32 + kb1 + e;
        int k2 = ks * 32 + kb2 + e;
        dst[e]     = cok ? (__bf16)B[(size_t)k1 * Ncols + col] : (__bf16)0.0f;
        dst[8 + e] = cok ? (__bf16)B[(size_t)k2 * Ncols + col] : (__bf16)0.0f;
    }
}

// ---------- fold w_gat with attention vectors: wa[k*4+h], wa[64+k*4+h] ----------
__global__ void k_prep(const float* __restrict__ wgat, const float* __restrict__ atts,
                       const float* __restrict__ attd, float* __restrict__ wa) {
    int t = threadIdx.x;
    if (t >= CIN * HEADS) return;
    int k = t >> 2, h = t & 3;
    float as = 0.f, ad = 0.f;
    for (int c = 0; c < HID; ++c) {
        float w = wgat[k * (HEADS * HID) + h * HID + c];
        as = fmaf(w, atts[h * HID + c], as);
        ad = fmaf(w, attd[h * HID + c], ad);
    }
    wa[k * 4 + h] = as;
    wa[64 + k * 4 + h] = ad;
}

// ---------- per (node, head): attention logits + init accumulators ----------
__global__ void k_node(const float* __restrict__ x, const float* __restrict__ wa,
                       float* __restrict__ asrc, float* __restrict__ adst,
                       unsigned* __restrict__ menc, float* __restrict__ ssum,
                       float* __restrict__ deg) {
    int i = blockIdx.x * blockDim.x + threadIdx.x;
    if (i >= GN * HEADS) return;
    int n = i >> 2, h = i & 3;
    float as = 0.f, ad = 0.f;
#pragma unroll
    for (int k = 0; k < CIN; ++k) {
        float xv = x[n * CIN + k];
        as = fmaf(xv, wa[k * 4 + h], as);
        ad = fmaf(xv, wa[64 + k * 4 + h], ad);
    }
    asrc[i] = as; adst[i] = ad;
    menc[i] = 0u; ssum[i] = 0.f;
    if (h == 0) deg[n] = 1.0f;   // self-loop weight
}

// ---------- edge pass 1: leaky-relu logits + segment max ----------
__global__ void k_edge_max(const int* __restrict__ src, const int* __restrict__ dst,
                           const float* __restrict__ asrc, const float* __restrict__ adst,
                           float* __restrict__ ebuf, unsigned* __restrict__ menc) {
    int i = blockIdx.x * blockDim.x + threadIdx.x;
    if (i >= GE + GN) return;
    int s = (i < GE) ? src[i] : (i - GE);
    int d = (i < GE) ? dst[i] : (i - GE);
#pragma unroll
    for (int h = 0; h < HEADS; ++h) {
        float e = asrc[s * 4 + h] + adst[d * 4 + h];
        e = (e > 0.f) ? e : 0.2f * e;
        ebuf[i * 4 + h] = e;
        atomicMax(menc + d * 4 + h, fenc(e));
    }
}

// ---------- edge pass 2: exp + segment sum; also weighted degree ----------
__global__ void k_edge_exp(const int* __restrict__ src, const int* __restrict__ dst,
                           const float* __restrict__ ew, float* __restrict__ ebuf,
                           const unsigned* __restrict__ menc, float* __restrict__ ssum,
                           float* __restrict__ deg) {
    int i = blockIdx.x * blockDim.x + threadIdx.x;
    if (i >= GE + GN) return;
    int d = (i < GE) ? dst[i] : (i - GE);
#pragma unroll
    for (int h = 0; h < HEADS; ++h) {
        float e = ebuf[i * 4 + h];
        float v = __expf(e - fdec(menc[d * 4 + h]));
        ebuf[i * 4 + h] = v;
        atomicAdd(ssum + d * 4 + h, v);
    }
    if (i < GE) atomicAdd(deg + d, ew[i]);
}

// ---------- degree -> rsqrt ----------
__global__ void k_dinv(const float* __restrict__ deg, float* __restrict__ dinv) {
    int n = blockIdx.x * blockDim.x + threadIdx.x;
    if (n >= GN) return;
    float dg = deg[n];
    dinv[n] = (dg > 0.f) ? rsqrtf(fmaxf(dg, 1e-12f)) : 0.f;
}

// ---------- GAT scatter: wave per EPW edges; recompute src features from x ----
__global__ __launch_bounds__(256) void k_gat_scatter(
        const int* __restrict__ src, const int* __restrict__ dst,
        const float* __restrict__ x, const float* __restrict__ wgat,
        const float* __restrict__ ebuf, const float* __restrict__ ssum,
        float* __restrict__ out) {
    __shared__ float lw[CIN * HEADS * HID];   // 30720 B
    for (int t = threadIdx.x; t < CIN * HEADS * HID; t += blockDim.x)
        lw[t] = wgat[t];
    __syncthreads();
    int lane = threadIdx.x & 31;
    int base = (blockIdx.x * (int)(blockDim.x >> 5) + (int)(threadIdx.x >> 5)) * EPW;
    for (int it = 0; it < EPW; ++it) {
        int widx = base + it;
        if (widx >= GE + GN) return;
        int s = (widx < GE) ? src[widx] : (widx - GE);
        int d = (widx < GE) ? dst[widx] : (widx - GE);
        float xv[CIN];
#pragma unroll
        for (int k = 0; k < CIN; ++k) xv[k] = x[s * CIN + k];
        float coef[HEADS];
#pragma unroll
        for (int h = 0; h < HEADS; ++h)
            coef[h] = ebuf[widx * 4 + h] / (ssum[d * 4 + h] + 1e-16f) * 0.25f;
#pragma unroll
        for (int j = 0; j < 4; ++j) {
            int c = lane + 32 * j;
            float acc = 0.f;
#pragma unroll
            for (int h = 0; h < HEADS; ++h) {
                float hv = 0.f;
#pragma unroll
                for (int k = 0; k < CIN; ++k)
                    hv = fmaf(xv[k], lw[k * (HEADS * HID) + h * HID + c], hv);
                acc = fmaf(coef[h], hv, acc);
            }
            atomicAdd(out + d * HID + c, acc);
        }
    }
}

// ---------- fused bias + BatchNorm(eval) + ReLU ----------
__global__ void k_post(const float* __restrict__ in, const float* __restrict__ bias,
                       const float* __restrict__ g, const float* __restrict__ b,
                       const float* __restrict__ m, const float* __restrict__ v,
                       float* __restrict__ out, int total) {
    int i = blockIdx.x * blockDim.x + threadIdx.x;
    if (i >= total) return;
    int c = i & (HID - 1);
    float xv = in[i] + bias[c];
    xv = g[c] * (xv - m[c]) * rsqrtf(v[c] + 1e-5f) + b[c];
    out[i] = fmaxf(xv, 0.f);
}

// ---------- GCN self-loop init: out = dinv^2 * in ----------
__global__ void k_aggr_init(const float* __restrict__ in, const float* __restrict__ dinv,
                            float* __restrict__ out, int total) {
    int i = blockIdx.x * blockDim.x + threadIdx.x;
    if (i >= total) return;
    float dv = dinv[i >> 7];
    out[i] = dv * dv * in[i];
}

// ---------- GCN edge scatter: wave per edge ----------
__global__ __launch_bounds__(256) void k_aggr_edge(
        const int* __restrict__ src, const int* __restrict__ dst,
        const float* __restrict__ ew, const float* __restrict__ dinv,
        const float* __restrict__ in, float* __restrict__ out) {
    int widx = blockIdx.x * (blockDim.x >> 5) + (threadIdx.x >> 5);
    int lane = threadIdx.x & 31;
    if (widx >= GE) return;
    int s = src[widx], d = dst[widx];
    float coef = dinv[s] * ew[widx] * dinv[d];
#pragma unroll
    for (int j = 0; j < 4; ++j) {
        int c = lane + 32 * j;
        atomicAdd(out + d * HID + c, coef * in[s * HID + c]);
    }
}

// ---------- WMMA bf16 GEMM with pre-swizzled B fragments ----------
// KSTEPS compile-time so B fragments stay in fixed VGPRs (no movrel).
// EPI: 0 = plain, 1 = bias+relu, 2 = bias only.
template<int KSTEPS, int EPI>
__global__ void k_gemm(const float* __restrict__ A, const __bf16* __restrict__ F,
                       const float* __restrict__ bias, float* __restrict__ out,
                       int M, int Ncols) {
    constexpr int K = KSTEPS * 32;
    int lane = threadIdx.x & 31;
    int wave = threadIdx.x >> 5;          // = ntile
    bool hi = lane >= 16;
    int ml = lane & 15;
    int col = wave * 16 + ml;
    int kb1 = hi ? 8 : 0, kb2 = hi ? 24 : 16;

    // B fragments: coalesced 32B-per-lane loads, pinned in registers
    v16bf bfrag[KSTEPS];
    const v16bf* Fp = (const v16bf*)F;
#pragma unroll
    for (int s = 0; s < KSTEPS; ++s)
        bfrag[s] = Fp[(wave * KSTEPS + s) * 32 + lane];

    bool cok = col < Ncols;
    float bv = (bias && cok) ? bias[col] : 0.0f;

    for (int m = 0; m < MPB; ++m) {
        int row0 = (blockIdx.x * MPB + m) * 16;
        if (row0 >= M) return;
        const float* arow = A + (size_t)(row0 + ml) * K;
        v8f acc = {};
#pragma unroll
        for (int s = 0; s < KSTEPS; ++s) {
            const float4* p1 = (const float4*)(arow + s * 32 + kb1);
            const float4* p2 = (const float4*)(arow + s * 32 + kb2);
            float4 x0 = p1[0], x1 = p1[1], x2 = p2[0], x3 = p2[1];
            v16bf a;
            a[0]  = (__bf16)x0.x; a[1]  = (__bf16)x0.y; a[2]  = (__bf16)x0.z; a[3]  = (__bf16)x0.w;
            a[4]  = (__bf16)x1.x; a[5]  = (__bf16)x1.y; a[6]  = (__bf16)x1.z; a[7]  = (__bf16)x1.w;
            a[8]  = (__bf16)x2.x; a[9]  = (__bf16)x2.y; a[10] = (__bf16)x2.z; a[11] = (__bf16)x2.w;
            a[12] = (__bf16)x3.x; a[13] = (__bf16)x3.y; a[14] = (__bf16)x3.z; a[15] = (__bf16)x3.w;
            acc = __builtin_amdgcn_wmma_f32_16x16x32_bf16(
                      false, a, false, bfrag[s], (short)0, acc, false, false);
        }
        if (cok) {
            int rbase = row0 + (hi ? 8 : 0);
#pragma unroll
            for (int r = 0; r < 8; ++r) {
                float v = acc[r] + bv;
                if (EPI == 1) v = fmaxf(v, 0.f);
                out[(size_t)(rbase + r) * Ncols + col] = v;
            }
        }
    }
}

// =======================================================================
extern "C" void kernel_launch(void* const* d_in, const int* in_sizes, int n_in,
                              void* d_out, int out_size, void* d_ws, size_t ws_size,
                              hipStream_t stream) {
    const float* x        = (const float*)d_in[0];
    const int*   ei       = (const int*)d_in[1];
    const int*   src      = ei;
    const int*   dst      = ei + GE;
    const float* ew       = (const float*)d_in[2];
    const float* w_gat    = (const float*)d_in[3];
    const float* att_src  = (const float*)d_in[4];
    const float* att_dst  = (const float*)d_in[5];
    const float* b_gat    = (const float*)d_in[6];
    const float* w2 = (const float*)d_in[7];  const float* b2 = (const float*)d_in[8];
    const float* w3 = (const float*)d_in[9];  const float* b3 = (const float*)d_in[10];
    const float* w4 = (const float*)d_in[11]; const float* b4 = (const float*)d_in[12];
    const float* bn_g = (const float*)d_in[13];
    const float* bn_b = (const float*)d_in[14];
    const float* bn_m = (const float*)d_in[15];
    const float* bn_v = (const float*)d_in[16];
    const float* w_shared = (const float*)d_in[17]; const float* b_shared = (const float*)d_in[18];
    const float* w_mask1  = (const float*)d_in[19]; const float* b_mask1  = (const float*)d_in[20];
    const float* w_mask2  = (const float*)d_in[21]; const float* b_mask2  = (const float*)d_in[22];
    const float* w_inst1  = (const float*)d_in[23]; const float* b_inst1  = (const float*)d_in[24];
    const float* w_inst2  = (const float*)d_in[25]; const float* b_inst2  = (const float*)d_in[26];
    const float* w_edge1  = (const float*)d_in[27]; const float* b_edge1  = (const float*)d_in[28];
    const float* w_edge2  = (const float*)d_in[29]; const float* b_edge2  = (const float*)d_in[30];
    float* outp = (float*)d_out;

    // ---- workspace carve-up ----
    char* W = (char*)d_ws;
    size_t off = 0;
    auto take = [&](size_t bytes) -> void* {
        void* p = W + off;
        off += (bytes + 255) & ~(size_t)255;
        return p;
    };
    float*    wa    = (float*)take(128 * sizeof(float));
    float*    asrc  = (float*)take((size_t)GN * 4 * sizeof(float));
    float*    adst  = (float*)take((size_t)GN * 4 * sizeof(float));
    unsigned* menc  = (unsigned*)take((size_t)GN * 4 * sizeof(unsigned));
    float*    ssum  = (float*)take((size_t)GN * 4 * sizeof(float));
    float*    ebuf  = (float*)take((size_t)(GE + GN) * 4 * sizeof(float));
    float*    deg   = (float*)take((size_t)GN * sizeof(float));
    float*    dinv  = (float*)take((size_t)GN * sizeof(float));
    // bf16 B-fragment buffers: [ntiles*ksteps*32*16] elements each
    auto ftake = [&](int ntiles, int ksteps) -> __bf16* {
        return (__bf16*)take((size_t)ntiles * ksteps * 32 * 16 * sizeof(__bf16));
    };
    __bf16* f_w2 = ftake(8, 4); __bf16* f_w3 = ftake(8, 4); __bf16* f_w4 = ftake(8, 4);
    __bf16* f_ws = ftake(8, 4);
    __bf16* f_m1 = ftake(4, 4); __bf16* f_i1 = ftake(4, 4); __bf16* f_e1 = ftake(4, 4);
    __bf16* f_m2 = ftake(1, 2); __bf16* f_i2 = ftake(1, 2); __bf16* f_e2 = ftake(1, 2);
    float*    bufA  = (float*)take((size_t)GN * HID * sizeof(float));
    float*    bufB  = (float*)take((size_t)GN * HID * sizeof(float));
    float*    bufC  = (float*)take((size_t)GN * HID * sizeof(float));

    const int NH  = GN * HID;               // 6.4M
    const int ET  = GE + GN;                // edges incl. self loops
    const int MT  = GN / 16;                // 3125 row tiles
    const int MB  = (MT + MPB - 1) / MPB;   // GEMM grid blocks
    dim3 b256(256);

    // ---- pre-swizzle all weight matrices into WMMA B-fragment layout ----
    k_bfrag<<<4, b256, 0, stream>>>(w2,       f_w2, 128, 128, 8);
    k_bfrag<<<4, b256, 0, stream>>>(w3,       f_w3, 128, 128, 8);
    k_bfrag<<<4, b256, 0, stream>>>(w4,       f_w4, 128, 128, 8);
    k_bfrag<<<4, b256, 0, stream>>>(w_shared, f_ws, 128, 128, 8);
    k_bfrag<<<2, b256, 0, stream>>>(w_mask1,  f_m1, 128, 64, 4);
    k_bfrag<<<2, b256, 0, stream>>>(w_inst1,  f_i1, 128, 64, 4);
    k_bfrag<<<2, b256, 0, stream>>>(w_edge1,  f_e1, 128, 64, 4);
    k_bfrag<<<1, dim3(64), 0, stream>>>(w_mask2, f_m2, 64, 2, 1);
    k_bfrag<<<1, dim3(64), 0, stream>>>(w_inst2, f_i2, 64, 2, 1);
    k_bfrag<<<1, dim3(64), 0, stream>>>(w_edge2, f_e2, 64, 1, 1);

    // ---- GAT ----
    k_prep<<<1, 64, 0, stream>>>(w_gat, att_src, att_dst, wa);
    k_node<<<(GN * 4 + 255) / 256, b256, 0, stream>>>(x, wa, asrc, adst, menc, ssum, deg);
    k_zero<<<(NH + 255) / 256, b256, 0, stream>>>(bufC, NH);
    k_edge_max<<<(ET + 255) / 256, b256, 0, stream>>>(src, dst, asrc, adst, ebuf, menc);
    k_edge_exp<<<(ET + 255) / 256, b256, 0, stream>>>(src, dst, ew, ebuf, menc, ssum, deg);
    k_dinv<<<(GN + 255) / 256, b256, 0, stream>>>(deg, dinv);
    k_gat_scatter<<<(ET + 8 * EPW - 1) / (8 * EPW), b256, 0, stream>>>(
        src, dst, x, w_gat, ebuf, ssum, bufC);
    k_post<<<(NH + 255) / 256, b256, 0, stream>>>(bufC, b_gat,
        bn_g + 0 * HID, bn_b + 0 * HID, bn_m + 0 * HID, bn_v + 0 * HID, bufA, NH);

    // ---- 3 GCN layers ----
    const __bf16* fwl[3] = {f_w2, f_w3, f_w4};
    const float*  bl[3]  = {b2, b3, b4};
    for (int l = 0; l < 3; ++l) {
        k_gemm<4, 0><<<MB, dim3(256), 0, stream>>>(bufA, fwl[l], nullptr, bufB, GN, HID);
        k_aggr_init<<<(NH + 255) / 256, b256, 0, stream>>>(bufB, dinv, bufC, NH);
        k_aggr_edge<<<(GE + 7) / 8, b256, 0, stream>>>(src, dst, ew, dinv, bufB, bufC);
        int r = l + 1;
        k_post<<<(NH + 255) / 256, b256, 0, stream>>>(bufC, bl[l],
            bn_g + r * HID, bn_b + r * HID, bn_m + r * HID, bn_v + r * HID, bufA, NH);
    }

    // ---- shared dense: hs = relu(h @ w_shared + b) ----
    k_gemm<4, 1><<<MB, dim3(256), 0, stream>>>(bufA, f_ws, b_shared, bufB, GN, HID);

    // ---- mask head: 128->64 relu, 64->2 ----
    k_gemm<4, 1><<<MB, dim3(128), 0, stream>>>(bufB, f_m1, b_mask1, bufC, GN, 64);
    k_gemm<2, 2><<<MB, dim3(32), 0, stream>>>(bufC, f_m2, b_mask2, outp, GN, 2);
    // ---- instance head ----
    k_gemm<4, 1><<<MB, dim3(128), 0, stream>>>(bufB, f_i1, b_inst1, bufC, GN, 64);
    k_gemm<2, 2><<<MB, dim3(32), 0, stream>>>(bufC, f_i2, b_inst2, outp + (size_t)GN * 2, GN, 2);
    // ---- edge head ----
    k_gemm<4, 1><<<MB, dim3(128), 0, stream>>>(bufB, f_e1, b_edge1, bufC, GN, 64);
    k_gemm<2, 2><<<MB, dim3(32), 0, stream>>>(bufC, f_e2, b_edge2, outp + (size_t)GN * 4, GN, 1);

    (void)in_sizes; (void)n_in; (void)out_size; (void)ws_size;
}
